// GAT_36610301231191
// MI455X (gfx1250) — matrix operand order repaired and checked by reading, not compiled, not run on hardware
//
#include <hip/hip_runtime.h>
#include <hip/hip_bf16.h>

// ---------------------------------------------------------------------------
// GAT (2 layers, 4 heads) + attention-MIL tail for MI455X (gfx1250, wave32).
//   0) transpose kernels : W -> WT (f32 [h][E][K]), Wh -> WhT (f16 [h][E][N])
//   1) wh_gemm_kernel    : Wh[h] = X @ W[h] via v_wmma_f32_16x16x32_f16
//                          (BM=128, BN templated, register double-buffered staging,
//                          B-operand fetch pipelined one n-tile ahead)
//   2) scores_kernel     : si/sj = Wh . a_i / a_j
//   3) rowstats_kernel   : per-row online (max, sum-exp) over masked leaky-relu scores
//   4) attagg_kernel     : out[h] = softmax(P) @ Wh[h]; BM=64 rows/block for 4x Wh
//                          L2-traffic reuse; waves tiled RG row-groups x CG col-groups
//   5) combine_kernel    : tanh(mean over heads), float4 wide
//   6) tail kernels      : fc1/fc2, column softmax over N, attention pooling, final head
// WMMA operand fetch = 2x ds_load_b128 per operand:
//   A (16x32 f16): lane m=l%16, halves [kb..kb+7],[16+kb..16+kb+7], kb=8*(l>=16)
//   B (32x16 f16): staged transposed (sBt[n][k]); halves [kb..kb+7],[16+kb..16+kb+7], kb=8*(l>=16)
// ---------------------------------------------------------------------------

typedef __attribute__((ext_vector_type(16))) _Float16 v16h;
typedef __attribute__((ext_vector_type(8)))  _Float16 v8h;
typedef __attribute__((ext_vector_type(4)))  _Float16 v4h;
typedef __attribute__((ext_vector_type(8)))  float    v8f;

#define SHUF16(lo, hi) __builtin_shufflevector(lo, hi, 0,1,2,3,4,5,6,7,8,9,10,11,12,13,14,15)

#define NNODES 4096
#define D_IN   2560
#define D_HID  256
#define D_OUT  64
#define HEADS  4
#define DENSE  16
#define ATTH   6
#define NCLS   5

#define LDSP   40   // LDS row stride in halves: 80 bytes, multiple of 16

__device__ __forceinline__ v8f wmma_f16(v16h a, v16h b, v8f c) {
  return __builtin_amdgcn_wmma_f32_16x16x32_f16(false, a, false, b, (short)0, c, false, false);
}
__device__ __forceinline__ v8f vzero8() {
  return (v8f){0.f,0.f,0.f,0.f,0.f,0.f,0.f,0.f};
}

// ---------------------------------------------------------------------------
// 0a) f32 transpose per head: in [R][C] -> out [C][R].  Tile 32x32, 256 thr.
// ---------------------------------------------------------------------------
__global__ void transpose32_kernel(const float* __restrict__ in, float* __restrict__ out,
                                   int R, int C)
{
  __shared__ alignas(16) float sT[32][36];
  const int hd = blockIdx.z;
  const float* ip = in  + (size_t)hd * R * C;
  float*       op = out + (size_t)hd * R * C;
  const int r0 = blockIdx.x * 32, c0 = blockIdx.y * 32;
  const int tid = threadIdx.x;          // 256
  {
    int r = tid >> 3, c4 = (tid & 7) << 2;
    float4 v = *(const float4*)&ip[(size_t)(r0 + r) * C + c0 + c4];
    sT[c4 + 0][r] = v.x; sT[c4 + 1][r] = v.y;
    sT[c4 + 2][r] = v.z; sT[c4 + 3][r] = v.w;
  }
  __syncthreads();
  {
    int c = tid >> 3, r4 = (tid & 7) << 2;
    float4 v = { sT[c][r4], sT[c][r4 + 1], sT[c][r4 + 2], sT[c][r4 + 3] };
    *(float4*)&op[(size_t)(c0 + c) * R + r0 + r4] = v;
  }
}

// ---------------------------------------------------------------------------
// 0b) f16 transpose per head: in [R][C] -> out [C][R].  Tile 64x64, 256 thr.
// ---------------------------------------------------------------------------
__global__ void transpose16_kernel(const _Float16* __restrict__ in, _Float16* __restrict__ out,
                                   int R, int C)
{
  __shared__ alignas(16) _Float16 sT[64][72];
  const int hd = blockIdx.z;
  const _Float16* ip = in  + (size_t)hd * R * C;
  _Float16*       op = out + (size_t)hd * R * C;
  const int r0 = blockIdx.x * 64, c0 = blockIdx.y * 64;
  const int tid = threadIdx.x;          // 256
#pragma unroll
  for (int i = 0; i < 2; ++i) {
    int idx = tid + (i << 8);
    int r = idx >> 3, c8 = (idx & 7) << 3;
    v8h v = *(const v8h*)&ip[(size_t)(r0 + r) * C + c0 + c8];
#pragma unroll
    for (int q = 0; q < 8; ++q) sT[c8 + q][r] = v[q];
  }
  __syncthreads();
#pragma unroll
  for (int i = 0; i < 2; ++i) {
    int idx = tid + (i << 8);
    int c = idx >> 3, r8 = (idx & 7) << 3;
    v8h v = *(const v8h*)&sT[c][r8];
    *(v8h*)&op[(size_t)(c0 + c) * R + r0 + r8] = v;
  }
}

// ---------------------------------------------------------------------------
// 1) Wh[h][M][E] = X[M][K] @ W[h].  WT = W transposed, f32 [h][E][K].
//    BM=128 (8 waves x 16 rows), BN templated (each wave owns all BN cols).
// ---------------------------------------------------------------------------
template<int K, int E, int BN>
__global__ void wh_gemm_kernel(const float* __restrict__ X,
                               const float* __restrict__ WT,
                               _Float16* __restrict__ Wh, int M)
{
  constexpr int NTG = BN / 16;          // n-tiles per wave
  constexpr int BRG = BN / 32;          // bReg count
  __shared__ alignas(16) _Float16 sA [128][LDSP];  // [m][k]
  __shared__ alignas(16) _Float16 sBt[ BN][LDSP];  // [n][k] (transposed)

  const int head   = blockIdx.z;
  const int bm0    = blockIdx.x * 128;
  const int bn0    = blockIdx.y * BN;
  const int tid    = threadIdx.x;       // 256
  const int lane   = tid & 31;
  const int w      = tid >> 5;
  const int lane15 = lane & 15;
  const int hi     = lane >> 4;

  const float* WTh = WT + (size_t)head * K * E;

  // staging assignment (compile-time strides)
  const int srow = tid >> 3;            // 0..31
  const int scol = (tid & 7) << 2;      // 0..28 step 4
  const float* Xs = X   + (size_t)(bm0 + srow) * K + scol;
  const float* Ws = WTh + (size_t)(bn0 + srow) * K + scol;

  v8f acc[NTG];
#pragma unroll
  for (int t = 0; t < NTG; ++t) acc[t] = vzero8();

  float4 aReg[4], bReg[BRG];
#pragma unroll
  for (int i = 0; i < 4; ++i) aReg[i] = *(const float4*)&Xs[(size_t)(i << 5) * K];
#pragma unroll
  for (int i = 0; i < BRG; ++i) bReg[i] = *(const float4*)&Ws[(size_t)(i << 5) * K];

  auto loadB = [&](int nt) -> v16h {
    v8h blo = *(const v8h*)&sBt[(nt << 4) + lane15][(hi << 3)];
    v8h bhi = *(const v8h*)&sBt[(nt << 4) + lane15][16 + (hi << 3)];
    return SHUF16(blo, bhi);
  };

  for (int kb = 0; kb < K; kb += 32) {
    // commit staged registers to LDS (f32 -> f16)
#pragma unroll
    for (int i = 0; i < 4; ++i) {
      v4h h4 = { (_Float16)aReg[i].x, (_Float16)aReg[i].y,
                 (_Float16)aReg[i].z, (_Float16)aReg[i].w };
      *(v4h*)&sA[srow + (i << 5)][scol] = h4;
    }
#pragma unroll
    for (int i = 0; i < BRG; ++i) {
      v4h h4 = { (_Float16)bReg[i].x, (_Float16)bReg[i].y,
                 (_Float16)bReg[i].z, (_Float16)bReg[i].w };
      *(v4h*)&sBt[srow + (i << 5)][scol] = h4;
    }
    __syncthreads();

    // prefetch next k-tile while WMMAs run
    if (kb + 32 < K) {
#pragma unroll
      for (int i = 0; i < 4; ++i) aReg[i] = *(const float4*)&Xs[(size_t)(i << 5) * K + kb + 32];
#pragma unroll
      for (int i = 0; i < BRG; ++i) bReg[i] = *(const float4*)&Ws[(size_t)(i << 5) * K + kb + 32];
    }

    v8h alo = *(const v8h*)&sA[(w << 4) + lane15][(hi << 3)];
    v8h ahi = *(const v8h*)&sA[(w << 4) + lane15][16 + (hi << 3)];
    v16h a = SHUF16(alo, ahi);

    // B-operand fetch pipelined one n-tile ahead of its WMMA
    v16h bcur = loadB(0);
#pragma unroll
    for (int nt = 0; nt < NTG; ++nt) {
      v16h bnext = (nt + 1 < NTG) ? loadB(nt + 1) : bcur;
      acc[nt] = wmma_f16(a, bcur, acc[nt]);
      bcur = bnext;
    }
    __syncthreads();
  }

#pragma unroll
  for (int nt = 0; nt < NTG; ++nt)
#pragma unroll
    for (int v = 0; v < 8; ++v) {
      int m = bm0 + (w << 4) + v + (hi << 3);
      int n = bn0 + (nt << 4) + lane15;
      Wh[((size_t)head * M + m) * E + n] = (_Float16)acc[nt][v];
    }
}

// ---------------------------------------------------------------------------
// 2) si[h][n] = Wh[h][n][:] . ai[h][:], sj likewise (v8h wide Wh reads).
// ---------------------------------------------------------------------------
__global__ void scores_kernel(const _Float16* __restrict__ Wh,
                              const float* __restrict__ ai,
                              const float* __restrict__ aj,
                              float* __restrict__ si, float* __restrict__ sj,
                              int Nn, int E)
{
  int g = blockIdx.x * blockDim.x + threadIdx.x;
  if (g >= HEADS * Nn) return;
  int h = g / Nn, n = g % Nn;
  const _Float16* row = Wh + ((size_t)h * Nn + n) * E;
  const float* aih = ai + h * E;
  const float* ajh = aj + h * E;
  float vi = 0.f, vj = 0.f;
  for (int e = 0; e < E; e += 8) {
    v8h x8 = *(const v8h*)&row[e];
#pragma unroll
    for (int q = 0; q < 8; ++q) {
      float x = (float)x8[q];
      vi += x * aih[e + q];
      vj += x * ajh[e + q];
    }
  }
  si[g] = vi;
  sj[g] = vj;
}

// ---------------------------------------------------------------------------
// 3) Per-row masked softmax stats, int4/float4 wide. One block per row.
// ---------------------------------------------------------------------------
__global__ void rowstats_kernel(const int* __restrict__ adj,
                                const float* __restrict__ si,
                                const float* __restrict__ sj,
                                float* __restrict__ rowm,
                                float* __restrict__ rowinvl, int Nn)
{
  const int i   = blockIdx.x;
  const int tid = threadIdx.x;

  float m[HEADS], l[HEADS], sv[HEADS];
#pragma unroll
  for (int h = 0; h < HEADS; ++h) { m[h] = -3.0e38f; l[h] = 0.f; sv[h] = si[h * Nn + i]; }

  const int* arow = adj + (size_t)i * Nn;
  for (int j = tid << 2; j < Nn; j += 1024) {
    __builtin_prefetch(arow + j + 2048, 0, 0);
    int4 a4 = *(const int4*)&arow[j];
    const int av[4] = { a4.x, a4.y, a4.z, a4.w };
    float sjv[HEADS][4];
#pragma unroll
    for (int h = 0; h < HEADS; ++h) {
      float4 s4 = *(const float4*)&sj[h * Nn + j];
      sjv[h][0] = s4.x; sjv[h][1] = s4.y; sjv[h][2] = s4.z; sjv[h][3] = s4.w;
    }
#pragma unroll
    for (int q = 0; q < 4; ++q) {
      if (av[q] > 0) {
#pragma unroll
        for (int h = 0; h < HEADS; ++h) {
          float z = sv[h] + sjv[h][q];
          float e = z > 0.f ? z : 0.01f * z;
          if (e > m[h]) { l[h] = l[h] * __expf(m[h] - e) + 1.f; m[h] = e; }
          else          { l[h] += __expf(e - m[h]); }
        }
      }
    }
  }

  __shared__ float sm[HEADS][256];
  __shared__ float sl[HEADS][256];
#pragma unroll
  for (int h = 0; h < HEADS; ++h) { sm[h][tid] = m[h]; sl[h][tid] = l[h]; }
  __syncthreads();
  for (int s = 128; s > 0; s >>= 1) {
    if (tid < s) {
#pragma unroll
      for (int h = 0; h < HEADS; ++h) {
        float mA = sm[h][tid],     lA = sl[h][tid];
        float mB = sm[h][tid + s], lB = sl[h][tid + s];
        float mx = fmaxf(mA, mB);
        sm[h][tid] = mx;
        sl[h][tid] = lA * __expf(mA - mx) + lB * __expf(mB - mx);
      }
    }
    __syncthreads();
  }
  if (tid == 0) {
#pragma unroll
    for (int h = 0; h < HEADS; ++h) {
      float L = sl[h][0];
      rowm[h * Nn + i]    = sm[h][0];
      rowinvl[h * Nn + i] = (L > 0.f) ? (1.f / L) : 0.f;
    }
  }
}

// ---------------------------------------------------------------------------
// 4) outh[h][i][:] = P @ Wh[h].  Block = (RG*16 rows) x 1 head; RG*CG waves:
//    wave (rg,cg) computes rows rg*16..+15, cols cg*NT*16..+NT*16.
// ---------------------------------------------------------------------------
template<int E, int RG, int CG>
__global__ void attagg_kernel(const int* __restrict__ adj,
                              const float* __restrict__ si,
                              const float* __restrict__ sj,
                              const float* __restrict__ rowm,
                              const float* __restrict__ rowinvl,
                              const _Float16* __restrict__ WhT,
                              float* __restrict__ outh, int Nn)
{
  constexpr int THREADS = RG * CG * 32;
  constexpr int NT      = E / (CG * 16);     // n-tiles per wave
  constexpr int BM      = RG * 16;           // rows per block
  constexpr int CH      = (E * 4) / THREADS; // v8h B chunks per thread
  constexpr int PS      = 4 / CG;            // P 4-col slots per thread

  __shared__ alignas(16) _Float16 sP [BM][LDSP];  // [i][j]
  __shared__ alignas(16) _Float16 sBt[ E][LDSP];  // [n][k] (transposed)

  const int head   = blockIdx.y;
  const int i0     = blockIdx.x * BM;
  const int tid    = threadIdx.x;
  const int lane   = tid & 31;
  const int w      = tid >> 5;
  const int rg     = w / CG;
  const int cg     = w % CG;
  const int lane15 = lane & 15;
  const int hi     = lane >> 4;

  const _Float16* WT = WhT + (size_t)head * E * Nn;

  v8f acc[NT];
#pragma unroll
  for (int t = 0; t < NT; ++t) acc[t] = vzero8();

  // P-slot ownership: slot s covers row pi[s], cols pj[s]..pj[s]+3
  int pi[PS], pj[PS];
  float p_si[PS], p_m[PS], p_il[PS];
  const int* arow[PS];
#pragma unroll
  for (int s = 0; s < PS; ++s) {
    int idx = tid + s * THREADS;
    pi[s] = idx >> 3;
    pj[s] = (idx & 7) << 2;
    p_si[s] = si[head * Nn + i0 + pi[s]];
    p_m [s] = rowm[head * Nn + i0 + pi[s]];
    p_il[s] = rowinvl[head * Nn + i0 + pi[s]];
    arow[s] = adj + (size_t)(i0 + pi[s]) * Nn + pj[s];
  }
  const float* sjh = sj + (size_t)head * Nn;

  // double-buffer registers
  v8h    breg[CH];
  int4   areg[PS];
  float4 sreg[PS];

  auto preload = [&](int j0) {
#pragma unroll
    for (int s = 0; s < PS; ++s) {
      areg[s] = *(const int4*)  &arow[s][j0];
      sreg[s] = *(const float4*)&sjh[j0 + pj[s]];
    }
#pragma unroll
    for (int i = 0; i < CH; ++i) {
      int idx  = tid + i * THREADS;
      int n    = idx >> 2;
      int koff = (idx & 3) << 3;
      breg[i] = *(const v8h*)&WT[(size_t)n * Nn + j0 + koff];
    }
  };
  preload(0);

  auto loadB = [&](int nt) -> v16h {
    int n = cg * (NT * 16) + (nt << 4) + lane15;
    v8h blo = *(const v8h*)&sBt[n][(hi << 3)];
    v8h bhi = *(const v8h*)&sBt[n][16 + (hi << 3)];
    return SHUF16(blo, bhi);
  };

  for (int j0 = 0; j0 < Nn; j0 += 32) {
    // P tile from staged regs: p = adj ? exp(lrelu(si+sj) - m) / l : 0
#pragma unroll
    for (int s = 0; s < PS; ++s) {
      const int   av[4]  = { areg[s].x, areg[s].y, areg[s].z, areg[s].w };
      const float svv[4] = { sreg[s].x, sreg[s].y, sreg[s].z, sreg[s].w };
      v4h p4;
#pragma unroll
      for (int q = 0; q < 4; ++q) {
        float p = 0.f;
        if (av[q] > 0) {
          float z = p_si[s] + svv[q];
          float e = z > 0.f ? z : 0.01f * z;
          p = __expf(e - p_m[s]) * p_il[s];
        }
        p4[q] = (_Float16)p;
      }
      *(v4h*)&sP[pi[s]][pj[s]] = p4;
    }
    // commit staged B chunks to LDS (contiguous 16B stores)
#pragma unroll
    for (int i = 0; i < CH; ++i) {
      int idx  = tid + i * THREADS;
      int n    = idx >> 2;
      int koff = (idx & 3) << 3;
      *(v8h*)&sBt[n][koff] = breg[i];
    }
    __syncthreads();

    if (j0 + 32 < Nn) preload(j0 + 32);

    v8h alo = *(const v8h*)&sP[(rg << 4) + lane15][(hi << 3)];
    v8h ahi = *(const v8h*)&sP[(rg << 4) + lane15][16 + (hi << 3)];
    v16h a = SHUF16(alo, ahi);

    // B-operand fetch pipelined one n-tile ahead of its WMMA
    v16h bcur = loadB(0);
#pragma unroll
    for (int nt = 0; nt < NT; ++nt) {
      v16h bnext = (nt + 1 < NT) ? loadB(nt + 1) : bcur;
      acc[nt] = wmma_f16(a, bcur, acc[nt]);
      bcur = bnext;
    }
    __syncthreads();
  }

#pragma unroll
  for (int nt = 0; nt < NT; ++nt)
#pragma unroll
    for (int v = 0; v < 8; ++v) {
      int m = i0 + (rg << 4) + v + (hi << 3);
      int n = cg * (NT * 16) + (nt << 4) + lane15;
      outh[((size_t)head * Nn + m) * E + n] = acc[nt][v];
    }
}

// ---------------------------------------------------------------------------
// 5) x = tanh(mean over heads), float4 wide
// ---------------------------------------------------------------------------
__global__ void combine_kernel(const float* __restrict__ outh,
                               float* __restrict__ x, int Nn, int E)
{
  int g = (blockIdx.x * blockDim.x + threadIdx.x) << 2;
  if (g >= Nn * E) return;
  float4 s = {0.f, 0.f, 0.f, 0.f};
#pragma unroll
  for (int h = 0; h < HEADS; ++h) {
    float4 v = *(const float4*)&outh[(size_t)h * Nn * E + g];
    s.x += v.x; s.y += v.y; s.z += v.z; s.w += v.w;
  }
  float4 r = { tanhf(s.x * (1.f / HEADS)), tanhf(s.y * (1.f / HEADS)),
               tanhf(s.z * (1.f / HEADS)), tanhf(s.w * (1.f / HEADS)) };
  *(float4*)&x[g] = r;
}

// ---------------------------------------------------------------------------
// 6a) s[n][t] = tanh(out2[n]@fc1+b1) @ fc2 + b2
// ---------------------------------------------------------------------------
__global__ void tail_fc_kernel(const float* __restrict__ out2,
                               const float* __restrict__ fc1w, const float* __restrict__ fc1b,
                               const float* __restrict__ fc2w, const float* __restrict__ fc2b,
                               float* __restrict__ s, int Nn)
{
  int n = blockIdx.x * blockDim.x + threadIdx.x;
  if (n >= Nn) return;
  const float* r = out2 + (size_t)n * D_OUT;
  float x[DENSE];
#pragma unroll
  for (int d = 0; d < DENSE; ++d) {
    float a = fc1b[d];
    for (int e = 0; e < D_OUT; ++e) a += r[e] * fc1w[e * DENSE + d];
    x[d] = tanhf(a);
  }
#pragma unroll
  for (int t = 0; t < ATTH; ++t) {
    float a = fc2b[t];
#pragma unroll
    for (int d = 0; d < DENSE; ++d) a += x[d] * fc2w[d * ATTH + t];
    s[n * ATTH + t] = a;
  }
}

// 6b) column softmax stats over N for each of the ATTH columns (one block)
__global__ void colstats_kernel(const float* __restrict__ s,
                                float* __restrict__ cmax, float* __restrict__ csum, int Nn)
{
  const int tid = threadIdx.x;
  float m[ATTH], l[ATTH];
#pragma unroll
  for (int t = 0; t < ATTH; ++t) { m[t] = -3.0e38f; l[t] = 0.f; }
  for (int n = tid; n < Nn; n += 256) {
#pragma unroll
    for (int t = 0; t < ATTH; ++t) {
      float e = s[n * ATTH + t];
      if (e > m[t]) { l[t] = l[t] * __expf(m[t] - e) + 1.f; m[t] = e; }
      else          { l[t] += __expf(e - m[t]); }
    }
  }
  __shared__ float sm[ATTH][256];
  __shared__ float sl[ATTH][256];
#pragma unroll
  for (int t = 0; t < ATTH; ++t) { sm[t][tid] = m[t]; sl[t][tid] = l[t]; }
  __syncthreads();
  for (int st = 128; st > 0; st >>= 1) {
    if (tid < st) {
#pragma unroll
      for (int t = 0; t < ATTH; ++t) {
        float mA = sm[t][tid], lA = sl[t][tid];
        float mB = sm[t][tid + st], lB = sl[t][tid + st];
        float mx = fmaxf(mA, mB);
        sm[t][tid] = mx;
        sl[t][tid] = lA * __expf(mA - mx) + lB * __expf(mB - mx);
      }
    }
    __syncthreads();
  }
  if (tid == 0) {
#pragma unroll
    for (int t = 0; t < ATTH; ++t) { cmax[t] = sm[t][0]; csum[t] = sl[t][0]; }
  }
}

// 6c) emb[t][d] = sum_n softmax_col(s)[n][t] * out2[n][d]   (one block, 384 thr)
__global__ void emb_kernel(const float* __restrict__ s,
                           const float* __restrict__ out2,
                           const float* __restrict__ cmax, const float* __restrict__ csum,
                           float* __restrict__ emb, int Nn)
{
  int t = threadIdx.x / D_OUT;
  int d = threadIdx.x % D_OUT;
  if (t >= ATTH) return;
  float mm = cmax[t];
  float inv = 1.f / csum[t];
  float acc = 0.f;
  for (int n = 0; n < Nn; ++n)
    acc += __expf(s[n * ATTH + t] - mm) * inv * out2[(size_t)n * D_OUT + d];
  emb[t * D_OUT + d] = acc;
}

// 6d) final head: avg over ATTH, sigmoid(fcf), argmax, softmax (one block, 64 thr)
__global__ void final_kernel(const float* __restrict__ emb,
                             const float* __restrict__ fcfw, const float* __restrict__ fcfb,
                             float* __restrict__ out)
{
  __shared__ float avg[D_OUT];
  __shared__ float lg[NCLS];
  int tid = threadIdx.x;
  if (tid < D_OUT) {
    float a = 0.f;
#pragma unroll
    for (int t = 0; t < ATTH; ++t) a += emb[t * D_OUT + tid];
    avg[tid] = a * (1.f / ATTH);
  }
  __syncthreads();
  if (tid < NCLS) {
    float a = fcfb[tid];
    for (int d = 0; d < D_OUT; ++d) a += avg[d] * fcfw[d * NCLS + tid];
    float v = 1.f / (1.f + __expf(-a));
    lg[tid] = v;
    out[tid] = v;                 // logits
  }
  __syncthreads();
  if (tid == 0) {
    int best = 0;
    float bv = lg[0];
    for (int c = 1; c < NCLS; ++c) if (lg[c] > bv) { bv = lg[c]; best = c; }
    out[NCLS] = (float)best;      // Y_hat
    float mx = bv, se = 0.f;
    float ex[NCLS];
    for (int c = 0; c < NCLS; ++c) { ex[c] = __expf(lg[c] - mx); se += ex[c]; }
    for (int c = 0; c < NCLS; ++c) out[NCLS + 1 + c] = ex[c] / se;  // Y_prob
  }
}

// ---------------------------------------------------------------------------
extern "C" void kernel_launch(void* const* d_in, const int* in_sizes, int n_in,
                              void* d_out, int out_size, void* d_ws, size_t ws_size,
                              hipStream_t stream)
{
  const float* h    = (const float*)d_in[0];
  const int*   adj  = (const int*)  d_in[1];
  const float* W1   = (const float*)d_in[2];
  const float* a1i  = (const float*)d_in[3];
  const float* a1j  = (const float*)d_in[4];
  const float* W2   = (const float*)d_in[5];
  const float* a2i  = (const float*)d_in[6];
  const float* a2j  = (const float*)d_in[7];
  const float* fc1w = (const float*)d_in[8];
  const float* fc1b = (const float*)d_in[9];
  const float* fc2w = (const float*)d_in[10];
  const float* fc2b = (const float*)d_in[11];
  const float* fcfw = (const float*)d_in[12];
  const float* fcfb = (const float*)d_in[13];
  float* out = (float*)d_out;

  char* ws = (char*)d_ws;
  size_t off = 0;
  auto alloc = [&](size_t bytes) -> void* {
    off = (off + 255) & ~(size_t)255;
    void* p = ws + off;
    off += bytes;
    return p;
  };

  const int N = NNODES;
  float*    W1T  = (float*)   alloc((size_t)HEADS * D_IN * D_HID * sizeof(float));
  float*    W2T  = (float*)   alloc((size_t)HEADS * D_HID * D_OUT * sizeof(float));
  _Float16* Wh1  = (_Float16*)alloc((size_t)HEADS * N * D_HID * sizeof(_Float16));
  _Float16* Wh1T = (_Float16*)alloc((size_t)HEADS * N * D_HID * sizeof(_Float16));
  _Float16* Wh2  = (_Float16*)alloc((size_t)HEADS * N * D_OUT * sizeof(_Float16));
  _Float16* Wh2T = (_Float16*)alloc((size_t)HEADS * N * D_OUT * sizeof(_Float16));
  float* si1  = (float*)alloc((size_t)HEADS * N * sizeof(float));
  float* sj1  = (float*)alloc((size_t)HEADS * N * sizeof(float));
  float* m1   = (float*)alloc((size_t)HEADS * N * sizeof(float));
  float* il1  = (float*)alloc((size_t)HEADS * N * sizeof(float));
  float* si2  = (float*)alloc((size_t)HEADS * N * sizeof(float));
  float* sj2  = (float*)alloc((size_t)HEADS * N * sizeof(float));
  float* m2   = (float*)alloc((size_t)HEADS * N * sizeof(float));
  float* il2  = (float*)alloc((size_t)HEADS * N * sizeof(float));
  float* outh1 = (float*)alloc((size_t)HEADS * N * D_HID * sizeof(float));
  float* x1    = (float*)alloc((size_t)N * D_HID * sizeof(float));
  float* outh2 = (float*)alloc((size_t)HEADS * N * D_OUT * sizeof(float));
  float* out2  = (float*)alloc((size_t)N * D_OUT * sizeof(float));
  float* sbuf  = (float*)alloc((size_t)N * ATTH * sizeof(float));
  float* cmax  = (float*)alloc(ATTH * sizeof(float));
  float* csum  = (float*)alloc(ATTH * sizeof(float));
  float* emb   = (float*)alloc(ATTH * D_OUT * sizeof(float));

  // ---- Layer 1 ----
  transpose32_kernel<<<dim3(D_IN / 32, D_HID / 32, HEADS), 256, 0, stream>>>(
      W1, W1T, D_IN, D_HID);
  wh_gemm_kernel<D_IN, D_HID, 128><<<dim3(N / 128, D_HID / 128, HEADS), 256, 0, stream>>>(
      h, W1T, Wh1, N);
  scores_kernel<<<(HEADS * N + 255) / 256, 256, 0, stream>>>(
      Wh1, a1i, a1j, si1, sj1, N, D_HID);
  transpose16_kernel<<<dim3(N / 64, D_HID / 64, HEADS), 256, 0, stream>>>(
      Wh1, Wh1T, N, D_HID);
  rowstats_kernel<<<N, 256, 0, stream>>>(adj, si1, sj1, m1, il1, N);
  attagg_kernel<D_HID, 4, 4><<<dim3(N / 64, HEADS), 512, 0, stream>>>(
      adj, si1, sj1, m1, il1, Wh1T, outh1, N);
  combine_kernel<<<(N * D_HID / 4 + 255) / 256, 256, 0, stream>>>(outh1, x1, N, D_HID);

  // ---- Layer 2 ----
  transpose32_kernel<<<dim3(D_HID / 32, D_OUT / 32, HEADS), 256, 0, stream>>>(
      W2, W2T, D_HID, D_OUT);
  wh_gemm_kernel<D_HID, D_OUT, 64><<<dim3(N / 128, D_OUT / 64, HEADS), 256, 0, stream>>>(
      x1, W2T, Wh2, N);
  scores_kernel<<<(HEADS * N + 255) / 256, 256, 0, stream>>>(
      Wh2, a2i, a2j, si2, sj2, N, D_OUT);
  transpose16_kernel<<<dim3(N / 64, D_OUT / 64, HEADS), 256, 0, stream>>>(
      Wh2, Wh2T, N, D_OUT);
  rowstats_kernel<<<N, 256, 0, stream>>>(adj, si2, sj2, m2, il2, N);
  attagg_kernel<D_OUT, 4, 2><<<dim3(N / 64, HEADS), 256, 0, stream>>>(
      adj, si2, sj2, m2, il2, Wh2T, outh2, N);
  combine_kernel<<<(N * D_OUT / 4 + 255) / 256, 256, 0, stream>>>(outh2, out2, N, D_OUT);

  // ---- Tail ----
  tail_fc_kernel<<<(N + 127) / 128, 128, 0, stream>>>(out2, fc1w, fc1b, fc2w, fc2b, sbuf, N);
  colstats_kernel<<<1, 256, 0, stream>>>(sbuf, cmax, csum, N);
  emb_kernel<<<1, ATTH * D_OUT, 0, stream>>>(sbuf, out2, cmax, csum, emb, N);
  final_kernel<<<1, 64, 0, stream>>>(emb, fcfw, fcfb, out);
}
